// ExpertParallel_46591805227028
// MI455X (gfx1250) — compile-verified, matrix operand order checked by reading, not run on hardware
//
#include <hip/hip_runtime.h>
#include <hip/hip_bf16.h>

// ---------------------------------------------------------------------------
// MoE expert-parallel FFN for MI455X (gfx1250, wave32, WMMA).
//   out[n,:] = (silu(x[n]@Wg[e]) * (x[n]@Wu[e])) @ Wd[e]^T,  e = idx[n]
// N=2048 tokens, E=16 experts, H=I=1024.
//
// Roofline: 12.9 GFLOP / ~210 MB HBM -> ~9us memory-bound floor, reachable
// only at bf16 WMMA rate (v_wmma_f32_16x16x32_bf16, fp32 accumulate).
// M-tile = 32 tokens/block so each expert's 12 MB of fp32 weights is streamed
// ~4x through L2 (vs 8x at M=16), and each B-fragment feeds 2 WMMA m-tiles.
// ---------------------------------------------------------------------------

typedef __bf16 bf16;
typedef __attribute__((ext_vector_type(16))) __bf16 v16bf;
typedef __attribute__((ext_vector_type(8)))  __bf16 v8bf;
typedef __attribute__((ext_vector_type(8)))  float  v8f;

#define N_TOK 2048
#define E_NUM 16
#define HDIM  1024
#define BLK_M 32
#define MAX_TILES 80    // <= N/BLK_M + E partial tiles

// ws layout (ints): [33] numTiles, [64..64+2048) perm,
// [4096..4096+MAX_TILES*4) tile descriptors {expert, start, cnt, pad}
#define WS_PERM  64
#define WS_TILES 4096

// ---------------------------------------------------------------------------
// Kernel 1: routing — histogram, prefix, scatter, tile list. Single block.
// ---------------------------------------------------------------------------
__global__ void moe_route(const int* __restrict__ idx, int* __restrict__ ws) {
  __shared__ int cnt[E_NUM];
  __shared__ int cur[E_NUM];
  const int t = threadIdx.x;
  if (t < E_NUM) cnt[t] = 0;
  __syncthreads();
  for (int n = t; n < N_TOK; n += blockDim.x) atomicAdd(&cnt[idx[n]], 1);
  __syncthreads();
  if (t == 0) {
    int o = 0, nt = 0;
    int* tiles = ws + WS_TILES;
    for (int e = 0; e < E_NUM; ++e) {
      cur[e] = o;
      const int c = cnt[e];
      for (int s = 0; s < c; s += BLK_M) {
        tiles[nt * 4 + 0] = e;
        tiles[nt * 4 + 1] = o + s;
        tiles[nt * 4 + 2] = (c - s) < BLK_M ? (c - s) : BLK_M;
        tiles[nt * 4 + 3] = 0;
        ++nt;
      }
      o += c;
    }
    for (int k = nt; k < MAX_TILES; ++k) tiles[k * 4 + 2] = 0;  // dead blocks
    ws[33] = nt;
  }
  __syncthreads();
  int* perm = ws + WS_PERM;
  for (int n = t; n < N_TOK; n += blockDim.x) {
    const int e = idx[n];
    const int p = atomicAdd(&cur[e], 1);
    perm[p] = n;  // order within expert does not affect per-token outputs
  }
}

// ---------------------------------------------------------------------------
// Fragment loaders (layouts per CDNA5 ISA 7.12.2, wave32).
// ---------------------------------------------------------------------------
__device__ __forceinline__ v8f vzero8() {
  v8f z;
#pragma unroll
  for (int i = 0; i < 8; ++i) z[i] = 0.f;
  return z;
}

// A-matrix 16x32 bf16 from LDS tile rows (row stride HDIM).
// lanes 0-15: M=lane, K = kb+0..7 then kb+16..23
// lanes16-31: M=lane-16, K = kb+8..15 then kb+24..31
__device__ __forceinline__ v16bf load_afrag(const bf16* __restrict__ rowbase,
                                            int kb, int lane) {
  const int m  = lane & 15;
  const int kh = (lane < 16) ? 0 : 8;
  const bf16* p = rowbase + m * HDIM + kb + kh;
  union { v16bf v; v8bf h[2]; } u;
  u.h[0] = *(const v8bf*)(p);
  u.h[1] = *(const v8bf*)(p + 16);
  return u.v;
}

// B-matrix 32x16 bf16 from row-major fp32 W[k][n] (gate/up: W[k*1024+n]).
// lane holds column n = n0+(lane&15); K chunks mirror the A layout.
__device__ __forceinline__ v16bf load_bfrag_rowmajor(const float* __restrict__ W,
                                                     int kb, int n0, int lane) {
  const int n  = n0 + (lane & 15);
  const int kh = (lane < 16) ? 0 : 8;
  const float* p = W + (size_t)(kb + kh) * HDIM + n;
  union { v16bf v; bf16 e[16]; } u;
#pragma unroll
  for (int j = 0; j < 8; ++j) u.e[j] = (bf16)p[j * HDIM];
  const float* q = p + 16 * HDIM;
#pragma unroll
  for (int j = 0; j < 8; ++j) u.e[8 + j] = (bf16)q[j * HDIM];
  return u.v;
}

// B-matrix 32x16 bf16 where B[k][n] = Wt[n*1024 + k] (down_proj contracts
// its LAST axis -> per-lane reads are contiguous fp32 runs).
__device__ __forceinline__ v16bf load_bfrag_colmajor(const float* __restrict__ Wt,
                                                     int kb, int n0, int lane) {
  const int n  = n0 + (lane & 15);
  const int kh = (lane < 16) ? 0 : 8;
  const float* p = Wt + (size_t)n * HDIM + kb + kh;
  union { v16bf v; bf16 e[16]; } u;
#pragma unroll
  for (int j = 0; j < 8; ++j) u.e[j] = (bf16)p[j];
#pragma unroll
  for (int j = 0; j < 8; ++j) u.e[8 + j] = (bf16)p[16 + j];
  return u.v;
}

// ---------------------------------------------------------------------------
// Kernel 2: fused up/gate/silu/down for one 32-token tile per block.
// 512 threads = 16 waves; wave w owns columns [w*64, w*64+64) (4 n-tiles),
// 2 WMMA m-tiles; each B-fragment is reused across both m-tiles.
// ---------------------------------------------------------------------------
__launch_bounds__(512)
__global__ void moe_ffn(const float* __restrict__ x,
                        const float* __restrict__ Wg,
                        const float* __restrict__ Wu,
                        const float* __restrict__ Wd,
                        const int* __restrict__ ws,
                        float* __restrict__ out) {
  __shared__ bf16 xA[BLK_M][HDIM];    // 64 KB  token tile
  __shared__ bf16 ubuf[BLK_M][HDIM];  // 64 KB  x@Wu (bf16)
  __shared__ bf16 ibuf[BLK_M][HDIM];  // 64 KB  silu(gate)*up (bf16)

  const int* tiles = ws + WS_TILES;
  const int bid = blockIdx.x;
  const int e   = tiles[bid * 4 + 0];
  const int st  = tiles[bid * 4 + 1];
  const int cnt = tiles[bid * 4 + 2];
  if (cnt == 0) return;
  const int* perm = ws + WS_PERM;

  const int tid  = threadIdx.x;
  const int lane = tid & 31;
  const int wave = tid >> 5;
  const int n0w  = wave * 64;

  // ---- stage x tile (fp32 -> bf16), zero-pad rows beyond cnt -------------
  for (int i = tid; i < BLK_M * (HDIM / 4); i += 512) {
    const int m  = i >> 8;            // 256 float4 per row
    const int c4 = (i & 255) << 2;
    float4 v = make_float4(0.f, 0.f, 0.f, 0.f);
    if (m < cnt)
      v = ((const float4*)(x + (size_t)perm[st + m] * HDIM))[c4 >> 2];
    xA[m][c4 + 0] = (bf16)v.x;
    xA[m][c4 + 1] = (bf16)v.y;
    xA[m][c4 + 2] = (bf16)v.z;
    xA[m][c4 + 3] = (bf16)v.w;
  }
  __syncthreads();

  const size_t eoff = (size_t)e * HDIM * HDIM;
  const int    mb   = (lane < 16) ? 0 : 8;   // C/D layout: M = mt*16+mb+r
  const int    nc   = lane & 15;             // C/D layout: N = n0 + (lane&15)

  v8f acc[2][4];

  // ---- pass 1: up = x @ Wu[e]  -> ubuf (bf16, wave-private columns) ------
  {
    const float* W = Wu + eoff;
#pragma unroll
    for (int mt = 0; mt < 2; ++mt)
#pragma unroll
      for (int nt = 0; nt < 4; ++nt) acc[mt][nt] = vzero8();
    for (int kb = 0; kb < HDIM; kb += 32) {
      const v16bf a0 = load_afrag(&xA[0][0],  kb, lane);
      const v16bf a1 = load_afrag(&xA[16][0], kb, lane);
#pragma unroll
      for (int nt = 0; nt < 4; ++nt) {
        const v16bf bF = load_bfrag_rowmajor(W, kb, n0w + nt * 16, lane);
        acc[0][nt] = __builtin_amdgcn_wmma_f32_16x16x32_bf16(
            false, a0, false, bF, (short)0, acc[0][nt], false, false);
        acc[1][nt] = __builtin_amdgcn_wmma_f32_16x16x32_bf16(
            false, a1, false, bF, (short)0, acc[1][nt], false, false);
      }
    }
#pragma unroll
    for (int mt = 0; mt < 2; ++mt)
#pragma unroll
      for (int nt = 0; nt < 4; ++nt) {
        const int n = n0w + nt * 16 + nc;
#pragma unroll
        for (int r = 0; r < 8; ++r)
          ubuf[mt * 16 + mb + r][n] = (bf16)acc[mt][nt][r];
      }
  }

  // ---- pass 2: gate = x @ Wg[e];  inter = silu(gate)*up -> ibuf ----------
  {
    const float* W = Wg + eoff;
#pragma unroll
    for (int mt = 0; mt < 2; ++mt)
#pragma unroll
      for (int nt = 0; nt < 4; ++nt) acc[mt][nt] = vzero8();
    for (int kb = 0; kb < HDIM; kb += 32) {
      const v16bf a0 = load_afrag(&xA[0][0],  kb, lane);
      const v16bf a1 = load_afrag(&xA[16][0], kb, lane);
#pragma unroll
      for (int nt = 0; nt < 4; ++nt) {
        const v16bf bF = load_bfrag_rowmajor(W, kb, n0w + nt * 16, lane);
        acc[0][nt] = __builtin_amdgcn_wmma_f32_16x16x32_bf16(
            false, a0, false, bF, (short)0, acc[0][nt], false, false);
        acc[1][nt] = __builtin_amdgcn_wmma_f32_16x16x32_bf16(
            false, a1, false, bF, (short)0, acc[1][nt], false, false);
      }
    }
#pragma unroll
    for (int mt = 0; mt < 2; ++mt)
#pragma unroll
      for (int nt = 0; nt < 4; ++nt) {
        const int n = n0w + nt * 16 + nc;
#pragma unroll
        for (int r = 0; r < 8; ++r) {
          const int row = mt * 16 + mb + r;
          const float g = acc[mt][nt][r];
          const float s = g / (1.f + __expf(-g));      // silu in fp32
          ibuf[row][n] = (bf16)(s * (float)ubuf[row][n]);
        }
      }
  }
  __syncthreads();  // pass 3 contracts over ALL intermediate columns

  // ---- pass 3: out[m,j] = sum_k inter[m,k] * Wd[e][j][k] -----------------
  {
    const float* W = Wd + eoff;
#pragma unroll
    for (int mt = 0; mt < 2; ++mt)
#pragma unroll
      for (int nt = 0; nt < 4; ++nt) acc[mt][nt] = vzero8();
    for (int kb = 0; kb < HDIM; kb += 32) {
      const v16bf a0 = load_afrag(&ibuf[0][0],  kb, lane);
      const v16bf a1 = load_afrag(&ibuf[16][0], kb, lane);
#pragma unroll
      for (int nt = 0; nt < 4; ++nt) {
        const v16bf bF = load_bfrag_colmajor(W, kb, n0w + nt * 16, lane);
        acc[0][nt] = __builtin_amdgcn_wmma_f32_16x16x32_bf16(
            false, a0, false, bF, (short)0, acc[0][nt], false, false);
        acc[1][nt] = __builtin_amdgcn_wmma_f32_16x16x32_bf16(
            false, a1, false, bF, (short)0, acc[1][nt], false, false);
      }
    }
#pragma unroll
    for (int mt = 0; mt < 2; ++mt)
#pragma unroll
      for (int nt = 0; nt < 4; ++nt) {
        const int n = n0w + nt * 16 + nc;
#pragma unroll
        for (int r = 0; r < 8; ++r) {
          const int m = mt * 16 + mb + r;
          if (m < cnt)
            out[(size_t)perm[st + m] * HDIM + n] = acc[mt][nt][r];
        }
      }
  }
}

// ---------------------------------------------------------------------------
extern "C" void kernel_launch(void* const* d_in, const int* in_sizes, int n_in,
                              void* d_out, int out_size, void* d_ws, size_t ws_size,
                              hipStream_t stream) {
  (void)in_sizes; (void)n_in; (void)out_size; (void)ws_size;
  const float* x   = (const float*)d_in[0];
  const int*   idx = (const int*)d_in[1];
  const float* Wg  = (const float*)d_in[2];
  const float* Wu  = (const float*)d_in[3];
  const float* Wd  = (const float*)d_in[4];
  float* out = (float*)d_out;
  int*   ws  = (int*)d_ws;

  hipLaunchKernelGGL(moe_route, dim3(1), dim3(256), 0, stream, idx, ws);
  hipLaunchKernelGGL(moe_ffn, dim3(MAX_TILES), dim3(512), 0, stream,
                     x, Wg, Wu, Wd, ws, out);
}
